// MimicLoss_76665166234031
// MI455X (gfx1250) — 8-bit matrix operand order repaired and checked by reading, compiled, not run on hardware
//
#include <hip/hip_runtime.h>
#include <hip/hip_bf16.h>


typedef int v8i  __attribute__((ext_vector_type(8)));
typedef int vs4i __attribute__((vector_size(16)));

#define NPRIORS 8192
#define NBATCH  32
#define NCH     256
#define HW      4096
#define CHW     (NCH * HW)
#define NBLK    64

#if defined(__HIP_DEVICE_COMPILE__) && __has_builtin(__builtin_amdgcn_global_load_async_to_lds_b128)
#define USE_ASYNC_LDS 1
#else
#define USE_ASYNC_LDS 0
#endif

__global__ __launch_bounds__(256) void prep_kernel(
    const float* __restrict__ priors, const int* __restrict__ labels,
    unsigned* __restrict__ bounds, unsigned char* __restrict__ posb,
    unsigned* __restrict__ msum)
{
    int i = blockIdx.x * blockDim.x + threadIdx.x;
    if (i < NPRIORS) {
        float cx = priors[i * 4 + 0];
        float cy = priors[i * 4 + 1];
        float bw = priors[i * 4 + 2];
        float bh = priors[i * 4 + 3];
        int xmin = (int)truncf((cx - bw * 0.5f) * 64.0f);
        int ymin = (int)truncf((cy - bh * 0.5f) * 64.0f);
        int xmax = (int)truncf((cx + bw * 0.5f) * 64.0f);
        int ymax = (int)truncf((cy + bh * 0.5f) * 64.0f);
        xmin = xmin < 0 ? 0 : xmin;
        ymin = ymin < 0 ? 0 : ymin;
        xmax = xmax > 64 ? 64 : xmax;
        ymax = ymax > 64 ? 64 : ymax;
        bounds[i] = (unsigned)xmin | ((unsigned)xmax << 8) |
                    ((unsigned)ymin << 16) | ((unsigned)ymax << 24);
    }
    if (i < NBATCH) msum[i] = 0u;
    if (i < NBATCH * NPRIORS) posb[i] = (labels[i] > 0) ? 1 : 0;
}

__device__ __forceinline__ v8i wmma_k_step(
    int kbase, const unsigned char* __restrict__ arow,
    const unsigned* __restrict__ bounds, int half, int x, int y, v8i acc)
{
    v8i a, bm;
#pragma unroll
    for (int v = 0; v < 8; ++v) {
        int abase = ((v >> 1) << 4) + ((v & 1) << 2) + (half << 3);
        a[v] = (int)*(const unsigned*)(arow + kbase + abase);
    }
#pragma unroll
    for (int v = 0; v < 8; ++v) {
        int kb = ((v >> 1) << 4) + ((v & 1) << 2) + (half << 3);
        uint4 q4 = *(const uint4*)(bounds + kbase + kb);
        unsigned r = 0;
#pragma unroll
        for (int j = 0; j < 4; ++j) {
            unsigned q = (j == 0) ? q4.x : (j == 1) ? q4.y : (j == 2) ? q4.z : q4.w;
            int xmin = (int)(q & 255u);
            int xmax = (int)((q >> 8) & 255u);
            int ymin = (int)((q >> 16) & 255u);
            int ymax = (int)(q >> 24);
            int tneg = (x - xmin) | (xmax - 1 - x) | (y - ymin) | (ymax - 1 - y);
            unsigned cov = (~(unsigned)tneg) >> 31;
            r |= cov << (8 * j);
        }
        bm[v] = (int)r;
    }
    return __builtin_amdgcn_wmma_i32_16x16x64_iu8(
         false, a,  false, bm, acc,
         false,  false);
}

__global__ __launch_bounds__(32) void mask_wmma_kernel(
    const unsigned* __restrict__ bounds, const unsigned char* __restrict__ posb,
    float* __restrict__ mask, unsigned* __restrict__ msum)
{
    const int lane = threadIdx.x;
    const int half = lane >> 4;
    const int l15  = lane & 15;
    const int n0   = blockIdx.x * 16;
    const int m0   = blockIdx.y * 16;
    const int px   = n0 + l15;
    const int x    = px & 63;
    const int y    = px >> 6;

    const unsigned char* arow = posb + (size_t)(m0 + l15) * NPRIORS;

    v8i acc0 = {0, 0, 0, 0, 0, 0, 0, 0};
    v8i acc1 = {0, 0, 0, 0, 0, 0, 0, 0};

    for (int k0 = 0; k0 < NPRIORS; k0 += 128) {
        acc0 = wmma_k_step(k0,      arow, bounds, half, x, y, acc0);
        acc1 = wmma_k_step(k0 + 64, arow, bounds, half, x, y, acc1);
    }
    v8i acc = acc0 + acc1;

#pragma unroll
    for (int r = 0; r < 8; ++r) {
        int b = m0 + r + (half << 3);
        unsigned hit = (unsigned)acc[r];
        mask[(size_t)b * HW + px] = hit ? 1.0f : 0.0f;
        if (hit) atomicAdd(&msum[b], 1u);
    }
}

__global__ __launch_bounds__(256) void reduce_kernel(
    const float* __restrict__ t, const float* __restrict__ s,
    const float* __restrict__ mask, float* __restrict__ partial)
{
    const int b   = blockIdx.y;
    const int blk = blockIdx.x;
    const int tid = threadIdx.x;

    __shared__ float smask[HW];
    const float* msrc = mask + (size_t)b * HW;

#if USE_ASYNC_LDS
    {
        vs4i __attribute__((address_space(1)))* g =
            (vs4i __attribute__((address_space(1)))*)(unsigned long long)(const void*)msrc;
        vs4i __attribute__((address_space(3)))* l =
            (vs4i __attribute__((address_space(3)))*)(unsigned)(unsigned long long)(void*)smask;
#pragma unroll
        for (int i = 0; i < 4; ++i) {
            int off4 = i * 256 + tid;
            __builtin_amdgcn_global_load_async_to_lds_b128(g + off4, l + off4, 0, 0);
        }
#if __has_builtin(__builtin_amdgcn_s_wait_asynccnt)
        __builtin_amdgcn_s_wait_asynccnt(0);
#else
        asm volatile("s_wait_asynccnt 0" ::: "memory");
#endif
    }
#else
#pragma unroll
    for (int i = 0; i < 4; ++i) {
        int off4 = i * 256 + tid;
        ((float4*)smask)[off4] = ((const float4*)msrc)[off4];
    }
#endif
    __syncthreads();

    const size_t base = (size_t)b * CHW + (size_t)blk * (CHW / NBLK);
    const float4* t4 = (const float4*)(t + base);
    const float4* s4 = (const float4*)(s + base);

    float acc = 0.0f;
#pragma unroll 4
    for (int i = 0; i < 16; ++i) {
        int off = i * 256 + tid;
        if (i + 4 < 16) {
            __builtin_prefetch(&t4[off + 1024], 0, 0);
            __builtin_prefetch(&s4[off + 1024], 0, 0);
        }
        float4 tv = t4[off];
        float4 sv = s4[off];
        float4 mv = *(const float4*)(smask + ((off * 4) & (HW - 1)));
        float dx = sv.x - tv.x, dy = sv.y - tv.y, dz = sv.z - tv.z, dw = sv.w - tv.w;
        acc = fmaf(dx * mv.x, dx, acc);
        acc = fmaf(dy * mv.y, dy, acc);
        acc = fmaf(dz * mv.z, dz, acc);
        acc = fmaf(dw * mv.w, dw, acc);
    }

    __shared__ float red[256];
    red[tid] = acc;
    __syncthreads();
#pragma unroll
    for (int st = 128; st > 0; st >>= 1) {
        if (tid < st) red[tid] += red[tid + st];
        __syncthreads();
    }
    if (tid == 0) partial[b * NBLK + blk] = red[0];
}

__global__ __launch_bounds__(32) void final_kernel(
    const float* __restrict__ partial, const unsigned* __restrict__ msum,
    float* __restrict__ out)
{
    int lane = threadIdx.x;
    float ssum = 0.0f;
    for (int j = 0; j < NBLK; ++j) ssum += partial[lane * NBLK + j];
    unsigned m = msum[lane];
    float safe = (m == 0u) ? 1.0f : (float)m;
    float contrib = ssum / safe / (float)NCH;

    __shared__ float    c[NBATCH];
    __shared__ unsigned mm[NBATCH];
    c[lane]  = contrib;
    mm[lane] = m;
    __syncthreads();
    if (lane == 0) {
        float loss = 0.0f;
        for (int b = 0; b < NBATCH; ++b)
            loss = (mm[b] == 0u) ? 0.0f : (loss + c[b]);
        out[0] = loss / (float)NBATCH;
    }
}

extern "C" void kernel_launch(void* const* d_in, const int* in_sizes, int n_in,
                              void* d_out, int out_size, void* d_ws, size_t ws_size,
                              hipStream_t stream)
{
    const float* t      = (const float*)d_in[0];
    const float* s      = (const float*)d_in[1];
    const float* priors = (const float*)d_in[2];
    const int*   labels = (const int*)d_in[3];
    (void)in_sizes; (void)n_in; (void)out_size; (void)ws_size;

    char* ws = (char*)d_ws;
    float*         mask    = (float*)(ws + 0);
    unsigned*      bounds  = (unsigned*)(ws + 524288);
    unsigned char* posb    = (unsigned char*)(ws + 557056);
    float*         partial = (float*)(ws + 819200);
    unsigned*      msum    = (unsigned*)(ws + 827392);

    prep_kernel<<<(NBATCH * NPRIORS) / 256, 256, 0, stream>>>(priors, labels, bounds, posb, msum);
    mask_wmma_kernel<<<dim3(HW / 16, NBATCH / 16), 32, 0, stream>>>(bounds, posb, mask, msum);
    reduce_kernel<<<dim3(NBLK, NBATCH), 256, 0, stream>>>(t, s, mask, partial);
    final_kernel<<<1, 32, 0, stream>>>(partial, msum, (float*)d_out);
}
